// GATATTMLPModule_63651415326971
// MI455X (gfx1250) — matrix operand order repaired and checked by reading, not compiled, not run on hardware
//
#include <hip/hip_runtime.h>

// ---------------------------------------------------------------------------
// Types for CDNA5 WMMA (wave32, 16x16x32 bf16 -> f32)
// ---------------------------------------------------------------------------
typedef __bf16 bf16;
typedef bf16  v16bf __attribute__((ext_vector_type(16)));
typedef bf16  v8bf  __attribute__((ext_vector_type(8)));
typedef bf16  v4bf  __attribute__((ext_vector_type(4)));
typedef float v8f   __attribute__((ext_vector_type(8)));

#define NN    4096
#define EG    131072
#define ERQ   65536
#define HH    4
#define LEAKY 0.2f

// two 16B ds_load_b128 -> one 16-element bf16 fragment
__device__ __forceinline__ v16bf ld_frag(const bf16* p) {
    v8bf lo = *(const v8bf*)p;
    v8bf hi = *(const v8bf*)(p + 8);
    return __builtin_shufflevector(lo, hi, 0, 1, 2, 3, 4, 5, 6, 7,
                                   8, 9, 10, 11, 12, 13, 14, 15);
}

// ---------------------------------------------------------------------------
// WMMA GEMM: C = [C +] act( act_in(A) @ W + bias )
// A: MxK f32 row-major, W: KxN f32 row-major, C: MxN f32.
// Block = 128 threads (4 waves). 64x64 C tile; wave w owns rows w*16..w*16+15
// with 4 wmma accumulators (a-fragment reused across 4 b-fragments).
// A staged in LDS in natural k order (lane half h reads k = 16h..16h+15) and B staged
// transposed, so every lane's fragment is 32 contiguous bytes (2x ds_load_b128).
// Main K body uses 128-bit global loads (all K call sites are %4==0; K=2 and
// the K=44 remainder go through the scalar zero-padded tail).
// ---------------------------------------------------------------------------
__global__ __launch_bounds__(128)
void gemm_bias_act(const float* __restrict__ A, const float* __restrict__ W,
                   const float* __restrict__ bias, float* __restrict__ C,
                   int M, int K, int N, int in_relu, int out_relu, int add_inplace)
{
    __shared__ bf16 As[64][40];   // 80B row stride: 16B aligned, bank-spread
    __shared__ bf16 Bt[64][40];   // transposed: Bt[n][k]
    const int tid  = threadIdx.x;
    const int lane = tid & 31;
    const int wave = tid >> 5;
    const int m0   = blockIdx.y * 64;
    const int n0   = blockIdx.x * 64;
    const int half = lane >> 4;
    const int l15  = lane & 15;
    const int row  = wave * 16 + l15;   // wave's fragment row in the 64-row tile

    v8f acc[4];
    #pragma unroll
    for (int s = 0; s < 4; ++s)
        acc[s] = (v8f){0.f, 0.f, 0.f, 0.f, 0.f, 0.f, 0.f, 0.f};

    for (int k0 = 0; k0 < K; k0 += 32) {
        if (k0 + 32 <= K) {
            // ---- unguarded staging, 128-bit loads (hot path) ----
            // A: 64 rows x 32 K = 512 float4; 4 per thread
            #pragma unroll
            for (int i = 0; i < 4; ++i) {
                int idx = tid + i * 128;
                int r = idx >> 3, c4 = (idx & 7) * 4;
                const float4* ap =
                    (const float4*)&A[(size_t)(m0 + r) * K + k0 + c4];
                float4 av = *ap;
                if (i == 0) __builtin_prefetch((const float*)ap + 32, 0, 1);
                if (in_relu) {
                    av.x = fmaxf(av.x, 0.f); av.y = fmaxf(av.y, 0.f);
                    av.z = fmaxf(av.z, 0.f); av.w = fmaxf(av.w, 0.f);
                }
                int g = c4 >> 3;
                int pc = c4;
                v4bf pk;
                pk[0] = (bf16)av.x; pk[1] = (bf16)av.y;
                pk[2] = (bf16)av.z; pk[3] = (bf16)av.w;
                *(v4bf*)&As[r][pc] = pk;   // 8B ds_store_b64
            }
            // B: 32 K x 64 N = 512 float4; 4 per thread; transposed scatter
            #pragma unroll
            for (int i = 0; i < 4; ++i) {
                int idx = tid + i * 128;
                int r = idx >> 4, c4 = (idx & 15) * 4;
                float4 bv = *(const float4*)&W[(size_t)(k0 + r) * N + n0 + c4];
                Bt[c4 + 0][r] = (bf16)bv.x;
                Bt[c4 + 1][r] = (bf16)bv.y;
                Bt[c4 + 2][r] = (bf16)bv.z;
                Bt[c4 + 3][r] = (bf16)bv.w;
            }
        } else {
            // ---- zero-padded scalar tail ----
            #pragma unroll
            for (int i = 0; i < 16; ++i) {
                int idx = tid + i * 128;
                int r = idx >> 5, c = idx & 31;
                int ka = k0 + c;
                float av = (ka < K) ? A[(size_t)(m0 + r) * K + ka] : 0.f;
                if (in_relu) av = fmaxf(av, 0.f);
                int pc = c;
                As[r][pc] = (bf16)av;
            }
            #pragma unroll
            for (int i = 0; i < 16; ++i) {
                int idx = tid + i * 128;
                int r = idx >> 6, c = idx & 63;
                int kb = k0 + r;
                Bt[c][r] = (bf16)((kb < K) ? W[(size_t)kb * N + n0 + c] : 0.f);
            }
        }
        __syncthreads();

        v16bf a = ld_frag(&As[row][half * 16]);
        #pragma unroll
        for (int s = 0; s < 4; ++s) {
            v16bf b = ld_frag(&Bt[s * 16 + l15][half * 16]);
            acc[s] = __builtin_amdgcn_wmma_f32_16x16x32_bf16(
                false, a, false, b, (short)0, acc[s], false, false);
        }
        __syncthreads();
    }

    // C/D layout: VGPR r -> row (r + 8*half), col = lane&15
    #pragma unroll
    for (int s = 0; s < 4; ++s) {
        int col = n0 + s * 16 + l15;
        #pragma unroll
        for (int r = 0; r < 8; ++r) {
            int rr = m0 + wave * 16 + r + 8 * half;
            float v = acc[s][r];
            if (bias) v += bias[col];
            if (out_relu) v = fmaxf(v, 0.f);
            size_t o = (size_t)rr * N + col;
            if (add_inplace) v += C[o];
            C[o] = v;
        }
    }
}

// ---------------------------------------------------------------------------
// Edge combine: f = leaky(xi[src] + xj[dst] + f + b); logits[e,h] = <f_h, attn_h>
// ---------------------------------------------------------------------------
__global__ __launch_bounds__(256)
void edge_combine_k(float* __restrict__ f, const float* __restrict__ xi,
                    const float* __restrict__ xj, const int* __restrict__ src,
                    const int* __restrict__ dst, const float* __restrict__ be,
                    const float* __restrict__ attn, float* __restrict__ logits,
                    int F, int H)
{
    __shared__ float part[8];
    int e = blockIdx.x, tid = threadIdx.x;
    if (tid < H) part[tid] = 0.f;
    __syncthreads();
    int s = src[e], d = dst[e];
    int D = F / H;
    for (int idx = tid; idx < F; idx += 256) {
        float v = xi[(size_t)s * F + idx] + xj[(size_t)d * F + idx] +
                  f[(size_t)e * F + idx] + be[idx];
        v = (v > 0.f) ? v : LEAKY * v;
        f[(size_t)e * F + idx] = v;
        int h = idx / D;
        atomicAdd(&part[h], v * attn[h * D + (idx % D)]);
    }
    __syncthreads();
    if (tid < H) logits[(size_t)e * H + tid] = part[tid];
}

// ---------------------------------------------------------------------------
// Segment softmax pieces (order-preserving uint encoding for float atomicMax)
// ---------------------------------------------------------------------------
__device__ __forceinline__ unsigned f2ord(float f) {
    unsigned u = __float_as_uint(f);
    return (u & 0x80000000u) ? ~u : (u | 0x80000000u);
}
__device__ __forceinline__ float ord2f(unsigned u) {
    return (u & 0x80000000u) ? __uint_as_float(u & 0x7fffffffu)
                             : __uint_as_float(~u);
}

__global__ void fill_f32_k(float* p, float v, int n) {
    int i = blockIdx.x * 256 + threadIdx.x;
    if (i < n) p[i] = v;
}
__global__ void fill_u32_k(unsigned* p, unsigned v, int n) {
    int i = blockIdx.x * 256 + threadIdx.x;
    if (i < n) p[i] = v;
}
__global__ void seg_max_k(const float* __restrict__ logits, const int* __restrict__ dst,
                          unsigned* __restrict__ mord, int E, int H) {
    int i = blockIdx.x * 256 + threadIdx.x;
    if (i >= E * H) return;
    int e = i / H, h = i % H;
    atomicMax(&mord[(size_t)dst[e] * H + h], f2ord(logits[i]));
}
__global__ void seg_expsum_k(float* __restrict__ logits, const int* __restrict__ dst,
                             const unsigned* __restrict__ mord, float* __restrict__ ssum,
                             int E, int H) {
    int i = blockIdx.x * 256 + threadIdx.x;
    if (i >= E * H) return;
    int e = i / H, h = i % H;
    float m = ord2f(mord[(size_t)dst[e] * H + h]);
    float w = __expf(logits[i] - m);
    logits[i] = w;
    atomicAdd(&ssum[(size_t)dst[e] * H + h], w);
}

// ---------------------------------------------------------------------------
// Aggregation: hout[dst] += hsrc[src] * (w / sum[dst]) per head
// ---------------------------------------------------------------------------
__global__ __launch_bounds__(256)
void aggregate_k(const float* __restrict__ hsrc, const float* __restrict__ w,
                 const float* __restrict__ ssum, const int* __restrict__ src,
                 const int* __restrict__ dst, float* __restrict__ hout, int F, int H)
{
    __shared__ float coef[8];
    int e = blockIdx.x, tid = threadIdx.x;
    int s = src[e], d = dst[e];
    if (tid < H) coef[tid] = w[(size_t)e * H + tid] / ssum[(size_t)d * H + tid];
    __syncthreads();
    int D = F / H;
    for (int idx = tid; idx < F; idx += 256)
        atomicAdd(&hout[(size_t)d * F + idx], hsrc[(size_t)s * F + idx] * coef[idx / D]);
}

// ---------------------------------------------------------------------------
// Flash-style MHA: 4 heads, head_dim 16, online softmax, LDS K/V tiles.
// grid = (NN/32, 4), block = 32 (one wave); one query row per thread.
// ---------------------------------------------------------------------------
__global__ __launch_bounds__(32)
void mha_k(const float* __restrict__ qkv, float* __restrict__ out, int Nn)
{
    __shared__ float kt[32][17], vt[32][17];
    int h = blockIdx.y;
    int tid = threadIdx.x;
    int row = blockIdx.x * 32 + tid;
    float q[16], acc[16];
    #pragma unroll
    for (int d = 0; d < 16; ++d) {
        q[d] = qkv[(size_t)row * 192 + h * 48 + d];
        acc[d] = 0.f;
    }
    float m = -1e30f, l = 0.f;
    for (int j0 = 0; j0 < Nn; j0 += 32) {
        int j = j0 + tid;
        #pragma unroll
        for (int d = 0; d < 16; ++d) {
            kt[tid][d] = qkv[(size_t)j * 192 + h * 48 + 16 + d];
            vt[tid][d] = qkv[(size_t)j * 192 + h * 48 + 32 + d];
        }
        __syncthreads();
        for (int jj = 0; jj < 32; ++jj) {
            float s = 0.f;
            #pragma unroll
            for (int d = 0; d < 16; ++d) s += q[d] * kt[jj][d];
            s *= 0.25f;  // 1/sqrt(16)
            float mn = fmaxf(m, s);
            float c = __expf(m - mn), p = __expf(s - mn);
            l = l * c + p;
            #pragma unroll
            for (int d = 0; d < 16; ++d) acc[d] = acc[d] * c + p * vt[jj][d];
            m = mn;
        }
        __syncthreads();
    }
    float inv = 1.f / l;
    #pragma unroll
    for (int d = 0; d < 16; ++d)
        out[(size_t)row * 64 + h * 16 + d] = acc[d] * inv;
}

__global__ void colmax_k(const float* __restrict__ x, float* __restrict__ feat, int Nn) {
    int c = threadIdx.x;
    float m = -1e30f;
    for (int r = 0; r < Nn; ++r) m = fmaxf(m, x[(size_t)r * 64 + c]);
    feat[c] = m;
}

__global__ void head_mlp_k(const float* __restrict__ feat,
                           const float* W1, const float* b1,
                           const float* W2, const float* b2,
                           const float* W3, const float* b3,
                           float* __restrict__ loc)
{
    __shared__ float x0[64], x1[64], x2[64];
    int t = threadIdx.x;
    x0[t] = feat[t]; __syncthreads();
    float a = b1[t];
    for (int k = 0; k < 64; ++k) a += x0[k] * W1[k * 64 + t];
    x1[t] = fmaxf(a, 0.f); __syncthreads();
    a = b2[t];
    for (int k = 0; k < 64; ++k) a += x1[k] * W2[k * 64 + t];
    x2[t] = fmaxf(a, 0.f); __syncthreads();
    if (t < 3) {
        a = b3[t];
        for (int k = 0; k < 64; ++k) a += x2[k] * W3[k * 3 + t];
        loc[t] = a;
    }
}

// ---------------------------------------------------------------------------
// Host-side orchestration
// ---------------------------------------------------------------------------
static inline void gemm(hipStream_t st, const float* A, const float* W, const float* bias,
                        float* C, int M, int K, int N, int in_relu, int out_relu, int addin) {
    dim3 g(N / 64, M / 64);
    hipLaunchKernelGGL(gemm_bias_act, g, dim3(128), 0, st,
                       A, W, bias, C, M, K, N, in_relu, out_relu, addin);
}
static inline void fillf(hipStream_t st, float* p, float v, int n) {
    hipLaunchKernelGGL(fill_f32_k, dim3((n + 255) / 256), dim3(256), 0, st, p, v, n);
}
static inline void fillu(hipStream_t st, unsigned* p, unsigned v, int n) {
    hipLaunchKernelGGL(fill_u32_k, dim3((n + 255) / 256), dim3(256), 0, st, p, v, n);
}

// one full EGAT conv: projections -> combine -> edge-softmax -> aggregate
static void run_egat(hipStream_t st,
                     const float* x, int relu_x, int xK,
                     const float* efeat, int relu_e, int eK,
                     const float* Wni, const float* Wnj, const float* Wfij,
                     const float* be, const float* attn, const float* Wnode,
                     const int* src, const int* dst, int E,
                     int Fe, int Fn,
                     float* xi, float* xj, float* fbuf, float* hsrc, float* hout,
                     float* logits, unsigned* mord, float* ssum)
{
    gemm(st, x, Wni, nullptr, xi, NN, xK, Fe, relu_x, 0, 0);
    gemm(st, x, Wnj, nullptr, xj, NN, xK, Fe, relu_x, 0, 0);
    gemm(st, efeat, Wfij, nullptr, fbuf, E, eK, Fe, relu_e, 0, 0);
    hipLaunchKernelGGL(edge_combine_k, dim3(E), dim3(256), 0, st,
                       fbuf, xi, xj, src, dst, be, attn, logits, Fe, HH);
    fillu(st, mord, 0x007FFFFFu /* ord(-inf) */, NN * HH);
    fillf(st, ssum, 0.f, NN * HH);
    hipLaunchKernelGGL(seg_max_k, dim3((E * HH + 255) / 256), dim3(256), 0, st,
                       logits, dst, mord, E, HH);
    hipLaunchKernelGGL(seg_expsum_k, dim3((E * HH + 255) / 256), dim3(256), 0, st,
                       logits, dst, mord, ssum, E, HH);
    gemm(st, x, Wnode, nullptr, hsrc, NN, xK, Fn, relu_x, 0, 0);
    fillf(st, hout, 0.f, NN * Fn);
    hipLaunchKernelGGL(aggregate_k, dim3(E), dim3(256), 0, st,
                       hsrc, logits, ssum, src, dst, hout, Fn, HH);
}

enum {
    c1_W_ni = 0, c1_W_nj, c1_W_fij, c1_attn, c1_W_node,
    c2_W_ni, c2_W_nj, c2_W_fij, c2_attn, c2_W_node,
    c3_W_ni, c3_W_nj, c3_W_fij, c3_attn, c3_W_node,
    W_np, W_el, W_l1, W_l2, W_l2a, W_l3, W_qkv, W_o, W_r1, W_r2, W_r3,
    c1_b_e, c2_b_e, c3_b_e, b_np, b_el, b_l1, b_l2, b_l2a, b_l3, b_qkv, b_o,
    b_r1, b_r2, b_r3, NPARAMS
};

extern "C" void kernel_launch(void* const* d_in, const int* in_sizes, int n_in,
                              void* d_out, int out_size, void* d_ws, size_t ws_size,
                              hipStream_t stream)
{
    const float* node_in = (const float*)d_in[0];
    const float* edge_in = (const float*)d_in[1];
    const int*   src_g   = (const int*)d_in[2];
    const int*   dst_g   = (const int*)d_in[3];
    const int*   src_rq  = (const int*)d_in[4];
    const int*   dst_rq  = (const int*)d_in[5];
    const float* P[NPARAMS];
    for (int i = 0; i < NPARAMS && (7 + i) < n_in; ++i)
        P[i] = (const float*)d_in[7 + i];

    // ---- workspace bump allocation (floats) ----
    float* ws = (float*)d_ws;
    size_t off = 0;
    float* nodeB  = ws + off; off += (size_t)NN * 64;
    float* edgeF  = ws + off; off += (size_t)EG * 64;
    float* elinkF = ws + off; off += (size_t)ERQ * 64;
    float* bufA   = ws + off; off += (size_t)EG * 512;   // c1 f (reused for c3 f)
    float* bufB   = ws + off; off += (size_t)EG * 256;   // c2 f
    float* xi     = ws + off; off += (size_t)NN * 512;
    float* xj     = ws + off; off += (size_t)NN * 512;
    float* hsrcB  = ws + off; off += (size_t)NN * 1024;  // c2 W_node output
    float* hA     = ws + off; off += (size_t)NN * 512;   // c1/c3 h
    float* hB     = ws + off; off += (size_t)NN * 1024;  // c2 h
    float* logits = ws + off; off += (size_t)EG * HH;
    unsigned* mord = (unsigned*)(ws + off); off += (size_t)NN * HH;
    float* ssum   = ws + off; off += (size_t)NN * HH;
    float* feat   = ws + off; off += 64;
    (void)ws_size; (void)in_sizes; (void)out_size;

    // ---- input projections ----
    gemm(stream, node_in, P[W_np], P[b_np], nodeB, NN, 44, 64, 0, 0, 0);
    gemm(stream, edge_in, P[W_el], P[b_el], edgeF, EG, 2, 64, 0, 0, 0);
    gemm(stream, edge_in + (size_t)EG * 2, P[W_el], P[b_el], elinkF, ERQ, 2, 64, 0, 0, 0);

    for (int layer = 0; layer < 2; ++layer) {
        // c1: x=node (NNx64), e=edgeF (EGx64) -> h in hA (NNx512), f in bufA (EGx512)
        run_egat(stream, nodeB, 0, 64, edgeF, 0, 64,
                 P[c1_W_ni], P[c1_W_nj], P[c1_W_fij], P[c1_b_e], P[c1_attn], P[c1_W_node],
                 src_g, dst_g, EG, 512, 512,
                 xi, xj, bufA, /*hsrc=*/xi, hA, logits, mord, ssum);

        // c2: x=relu(hA), e=relu(bufA) -> h in hB (NNx1024), f in bufB (EGx256)
        run_egat(stream, hA, 1, 512, bufA, 1, 512,
                 P[c2_W_ni], P[c2_W_nj], P[c2_W_fij], P[c2_b_e], P[c2_attn], P[c2_W_node],
                 src_g, dst_g, EG, 256, 1024,
                 xi, xj, bufB, hsrcB, hB, logits, mord, ssum);

        // edge = relu(bufB @ W_l1 + b_l1); node = relu(hB @ W_l2 + b_l2)
        gemm(stream, bufB, P[W_l1], P[b_l1], edgeF, EG, 256, 64, 0, 1, 0);
        gemm(stream, hB, P[W_l2], P[b_l2], nodeB, NN, 1024, 64, 0, 1, 0);

        // c3 on ref/query graph: x=node, e=elinkF -> h in hA, f in bufA (ERQx512)
        run_egat(stream, nodeB, 0, 64, elinkF, 0, 64,
                 P[c3_W_ni], P[c3_W_nj], P[c3_W_fij], P[c3_b_e], P[c3_attn], P[c3_W_node],
                 src_rq, dst_rq, ERQ, 512, 512,
                 xi, xj, bufA, /*hsrc=*/xi, hA, logits, mord, ssum);

        // edge_link = relu(bufA @ W_l3 + b_l3); node += relu(hA @ W_l2a + b_l2a)
        gemm(stream, bufA, P[W_l3], P[b_l3], elinkF, ERQ, 512, 64, 0, 1, 0);
        gemm(stream, hA, P[W_l2a], P[b_l2a], nodeB, NN, 512, 64, 0, 1, 1 /*residual*/);
    }

    // ---- MHA ----
    float* outNode = (float*)d_out + (size_t)EG * 64;
    gemm(stream, nodeB, P[W_qkv], P[b_qkv], xi, NN, 64, 192, 0, 0, 0);  // qkv -> xi
    hipLaunchKernelGGL(mha_k, dim3(NN / 32, HH), dim3(32), 0, stream, xi, xj, NN);
    gemm(stream, xj, P[W_o], P[b_o], outNode, NN, 64, 64, 0, 0, 0);

    // ---- edge output copy, pooled head ----
    hipMemcpyAsync(d_out, edgeF, (size_t)EG * 64 * sizeof(float),
                   hipMemcpyDeviceToDevice, stream);
    hipLaunchKernelGGL(colmax_k, dim3(1), dim3(64), 0, stream, outNode, feat, NN);
    hipLaunchKernelGGL(head_mlp_k, dim3(1), dim3(64), 0, stream,
                       feat, P[W_r1], P[b_r1], P[W_r2], P[b_r2], P[W_r3], P[b_r3],
                       (float*)d_out + (size_t)EG * 64 + (size_t)NN * 64);
}
